// FraudGNN_28355374088945
// MI455X (gfx1250) — matrix operand order repaired and checked by reading, not compiled, not run on hardware
//
#include <hip/hip_runtime.h>
#include <hip/hip_bf16.h>
#include <math.h>

typedef __attribute__((ext_vector_type(16))) _Float16 v16h;
typedef __attribute__((ext_vector_type(8)))  _Float16 v8h;
typedef __attribute__((ext_vector_type(8)))  float    v8f;

#define HID   128
#define HEADS 4
#define DH    32
#define BN_EPS 1e-5f

// ---------------------------------------------------------------------------
// Fused GEMM via v_wmma_f32_16x16x32_f16 (wave32).
// C[m,n] = act( sum_k A[m,k]*B[k,n] + bias[n] + addend[m,n] )
// REQUIREMENT: K and N multiples of 8 (true for every call: 16/32/64/128).
// Block = 256 threads = 8 waves; block tile 64(M) x 64(N); each wave owns a
// 16x32 strip (two 16x16 accumulators sharing one A fragment).
// Staging is fully branchless: addresses clamped in-bounds, validity applied
// via select -> two unconditional global_load_b128 per thread per tile.
// A staged row-major f16; B staged transposed (Bt[n][k]); LDS rows 96B so
// fragment reads are aligned ds_load_b128s.
// ---------------------------------------------------------------------------
__global__ __launch_bounds__(256)
void gemm_wmma_kernel(const float* __restrict__ A, int lda,
                      const float* __restrict__ B, int ldb,
                      const float* __restrict__ bias,
                      const float* __restrict__ addend,
                      float* __restrict__ C, int ldc,
                      int M, int N, int K, int act)
{
    __shared__ __attribute__((aligned(32))) _Float16 As[64][48];
    __shared__ __attribute__((aligned(32))) _Float16 Bt[64][48];
    const int m0 = blockIdx.x * 64, n0 = blockIdx.y * 64;
    const int t = threadIdx.x;
    const int lane = t & 31, wave = t >> 5;
    const int wm = (wave >> 1) * 16;   // 0,16,32,48
    const int wn = (wave & 1) * 32;    // 0,32
    const int r16 = lane & 15, hi = lane >> 4;

    // A staging: row rA = t>>2, 8-col segment sA; B staging: k-row rB = t>>3
    const int rA = t >> 2, sA = (t & 3) * 8;
    const int rB = t >> 3, sB = (t & 7) * 8;
    const int gmA  = m0 + rA;
    const int gmAc = gmA < M ? gmA : (M - 1);
    const bool mokA = (gmA < M);
    const int nnB  = n0 + sB;
    const bool nokB = (nnB + 8 <= N);
    const int nnBc = nokB ? nnB : 0;

    v8f acc0 = {}, acc1 = {};

    for (int k0 = 0; k0 < K; k0 += 32) {
        // ---- stage A tile (64 x 32), branchless ----
        {
            int kk = k0 + sA;
            bool ok = mokA && (kk + 8 <= K);
            const float* ap = A + (size_t)gmAc * (size_t)lda + (ok ? kk : 0);
            float4 x0 = *(const float4*)ap;
            float4 x1 = *(const float4*)(ap + 4);
            As[rA][sA + 0] = (_Float16)(ok ? x0.x : 0.f);
            As[rA][sA + 1] = (_Float16)(ok ? x0.y : 0.f);
            As[rA][sA + 2] = (_Float16)(ok ? x0.z : 0.f);
            As[rA][sA + 3] = (_Float16)(ok ? x0.w : 0.f);
            As[rA][sA + 4] = (_Float16)(ok ? x1.x : 0.f);
            As[rA][sA + 5] = (_Float16)(ok ? x1.y : 0.f);
            As[rA][sA + 6] = (_Float16)(ok ? x1.z : 0.f);
            As[rA][sA + 7] = (_Float16)(ok ? x1.w : 0.f);
        }
        // ---- stage B tile (32 x 64) transposed, branchless ----
        {
            int gk = k0 + rB;
            bool ok = nokB && (gk < K);
            int gkc = ok ? gk : 0;
            const float* bp = B + (size_t)gkc * (size_t)ldb + nnBc;
            float4 x0 = *(const float4*)bp;
            float4 x1 = *(const float4*)(bp + 4);
            Bt[sB + 0][rB] = (_Float16)(ok ? x0.x : 0.f);
            Bt[sB + 1][rB] = (_Float16)(ok ? x0.y : 0.f);
            Bt[sB + 2][rB] = (_Float16)(ok ? x0.z : 0.f);
            Bt[sB + 3][rB] = (_Float16)(ok ? x0.w : 0.f);
            Bt[sB + 4][rB] = (_Float16)(ok ? x1.x : 0.f);
            Bt[sB + 5][rB] = (_Float16)(ok ? x1.y : 0.f);
            Bt[sB + 6][rB] = (_Float16)(ok ? x1.z : 0.f);
            Bt[sB + 7][rB] = (_Float16)(ok ? x1.w : 0.f);
        }
        __syncthreads();
        // A fragment (16-bit A 16x32 layout): lanes0-15 K{0..7,16..23},
        // lanes16-31 K{8..15,24..31} -> two aligned 16B LDS vector loads
        const _Float16* ar = &As[wm + r16][0];
        v8h alo = *(const v8h*)(ar + hi * 8);
        v8h ahi = *(const v8h*)(ar + hi * 8 + 16);
        v16h a = __builtin_shufflevector(alo, ahi,
                 0, 1, 2, 3, 4, 5, 6, 7, 8, 9, 10, 11, 12, 13, 14, 15);
        // B fragments (16-bit B 32x16 layout): lane=col, lanes0-15 K{0..7,16..23},
        // lanes16-31 K{8..15,24..31} -> aligned 16B LDS vector loads
        const _Float16* br0 = &Bt[wn + r16][0];
        const _Float16* br1 = &Bt[wn + 16 + r16][0];
        v16h b0 = __builtin_shufflevector(*(const v8h*)(br0 + hi * 8), *(const v8h*)(br0 + hi * 8 + 16),
                 0, 1, 2, 3, 4, 5, 6, 7, 8, 9, 10, 11, 12, 13, 14, 15);
        v16h b1 = __builtin_shufflevector(*(const v8h*)(br1 + hi * 8), *(const v8h*)(br1 + hi * 8 + 16),
                 0, 1, 2, 3, 4, 5, 6, 7, 8, 9, 10, 11, 12, 13, 14, 15);
        acc0 = __builtin_amdgcn_wmma_f32_16x16x32_f16(false, a, false, b0,
                                                      (short)0, acc0, false, false);
        acc1 = __builtin_amdgcn_wmma_f32_16x16x32_f16(false, a, false, b1,
                                                      (short)0, acc1, false, false);
        __syncthreads();
    }
    // C/D layout: VGPR r -> M = r + 8*(lane>=16); N = lane&15
#pragma unroll
    for (int r = 0; r < 8; ++r) {
        int m = m0 + wm + r + hi * 8;
        if (m >= M) continue;
        int n1 = n0 + wn + r16;
        int n2 = n1 + 16;
        if (n1 < N) {
            float v = acc0[r];
            if (bias)   v += bias[n1];
            if (addend) v += addend[(size_t)m * ldc + n1];
            if (act == 1) v = fmaxf(v, 0.f);
            C[(size_t)m * ldc + n1] = v;
        }
        if (n2 < N) {
            float v = acc1[r];
            if (bias)   v += bias[n2];
            if (addend) v += addend[(size_t)m * ldc + n2];
            if (act == 1) v = fmaxf(v, 0.f);
            C[(size_t)m * ldc + n2] = v;
        }
    }
}

// ---------------------------------------------------------------------------
// SAGE helpers
// ---------------------------------------------------------------------------
__global__ void scatter_add_kernel(const float* __restrict__ h,
                                   const int* __restrict__ rows,
                                   const int* __restrict__ cols,
                                   float* __restrict__ msg, int E)
{
    long long t = (long long)blockIdx.x * blockDim.x + threadIdx.x;
    if (t >= (long long)E * HID) return;
    int e = (int)(t >> 7), f = (int)(t & 127);
    atomicAdd(&msg[(size_t)cols[e] * HID + f], h[(size_t)rows[e] * HID + f]);
}

__global__ void count_kernel(const int* __restrict__ cols, float* __restrict__ cnt, int E)
{
    int t = blockIdx.x * blockDim.x + threadIdx.x;
    if (t < E) atomicAdd(&cnt[cols[t]], 1.0f);
}

__global__ void div_cnt_kernel(float* __restrict__ msg, const float* __restrict__ cnt, int Nd)
{
    int t = blockIdx.x * blockDim.x + threadIdx.x;
    if (t >= Nd * HID) return;
    msg[t] /= fmaxf(cnt[t >> 7], 1.0f);
}

// agg[row] += o[row] / max(||o[row]||, 1e-12); one wave32 per row
__global__ __launch_bounds__(128)
void rownorm_acc_kernel(const float* __restrict__ o, float* __restrict__ agg, int rows)
{
    int lane = threadIdx.x & 31;
    int row  = blockIdx.x * 4 + (threadIdx.x >> 5);
    if (row >= rows) return;
    const float* op = o + (size_t)row * HID;
    float v0 = op[lane], v1 = op[lane + 32], v2 = op[lane + 64], v3 = op[lane + 96];
    float ss = v0 * v0 + v1 * v1 + v2 * v2 + v3 * v3;
    for (int d = 16; d >= 1; d >>= 1) ss += __shfl_xor(ss, d, 32);
    float inv = 1.0f / fmaxf(sqrtf(ss), 1e-12f);
    float* ap = agg + (size_t)row * HID;
    ap[lane]      += v0 * inv;
    ap[lane + 32] += v1 * inv;
    ap[lane + 64] += v2 * inv;
    ap[lane + 96] += v3 * inv;
}

// ---------------------------------------------------------------------------
// BatchNorm (training-mode, biased variance)
// ---------------------------------------------------------------------------
__global__ __launch_bounds__(256)
void bn_stats_kernel(const float* __restrict__ x, int M,
                     float* __restrict__ mean, float* __restrict__ var)
{
    __shared__ float s1[256], s2[256];
    int f = blockIdx.x, t = threadIdx.x;
    float a = 0.f, b = 0.f;
    for (int i = t; i < M; i += 256) {
        float v = x[(size_t)i * HID + f];
        a += v; b += v * v;
    }
    s1[t] = a; s2[t] = b; __syncthreads();
    for (int d = 128; d >= 1; d >>= 1) {
        if (t < d) { s1[t] += s1[t + d]; s2[t] += s2[t + d]; }
        __syncthreads();
    }
    if (t == 0) {
        float mu = s1[0] / M;
        mean[f] = mu;
        var[f]  = s2[0] / M - mu * mu;
    }
}

__global__ void bn_apply_relu_kernel(const float* __restrict__ x,
                                     const float* __restrict__ mean,
                                     const float* __restrict__ var,
                                     const float* __restrict__ g,
                                     const float* __restrict__ b,
                                     float* __restrict__ y, int M)
{
    long long t = (long long)blockIdx.x * blockDim.x + threadIdx.x;
    if (t >= (long long)M * HID) return;
    int f = (int)(t & 127);
    float v = (x[t] - mean[f]) * rsqrtf(var[f] + BN_EPS) * g[f] + b[f];
    y[t] = fmaxf(v, 0.f);
}

// ---------------------------------------------------------------------------
// HGT segment softmax (3 passes) with order-preserving uint float atomicMax
// ---------------------------------------------------------------------------
__device__ __forceinline__ unsigned enc_f(float f) {
    unsigned u = __float_as_uint(f);
    return (u & 0x80000000u) ? ~u : (u | 0x80000000u);
}
__device__ __forceinline__ float dec_f(unsigned u) {
    u = (u & 0x80000000u) ? (u & 0x7fffffffu) : ~u;
    return __uint_as_float(u);
}

__global__ void init_smax_kernel(unsigned* __restrict__ smax, int n)
{
    int t = blockIdx.x * blockDim.x + threadIdx.x;
    if (t < n) smax[t] = 0x007FFFFFu; // enc(-inf)
}

__global__ void hgt_score_kernel(const float* __restrict__ q, const float* __restrict__ ka,
                                 const int* __restrict__ rows, const int* __restrict__ cols,
                                 const float* __restrict__ prel,
                                 float* __restrict__ s, unsigned* __restrict__ smax, int E)
{
    int t = blockIdx.x * blockDim.x + threadIdx.x;
    if (t >= E * HEADS) return;
    int e = t >> 2, hh = t & 3;
    int r = rows[e], c = cols[e];
    const float* qp = q  + (size_t)c * HID + hh * DH;
    const float* kp = ka + (size_t)r * HID + hh * DH;
    float dot = 0.f;
#pragma unroll
    for (int d = 0; d < DH; ++d) dot += qp[d] * kp[d];
    float sv = dot * prel[hh] * 0.17677669529663689f; // 1/sqrt(32)
    s[t] = sv;
    atomicMax(&smax[(size_t)c * HEADS + hh], enc_f(sv));
}

__global__ void hgt_expsum_kernel(float* __restrict__ s, const unsigned* __restrict__ smax,
                                  const int* __restrict__ cols, float* __restrict__ den, int E)
{
    int t = blockIdx.x * blockDim.x + threadIdx.x;
    if (t >= E * HEADS) return;
    int e = t >> 2, hh = t & 3;
    int c = cols[e];
    float ex = expf(s[t] - dec_f(smax[(size_t)c * HEADS + hh]));
    s[t] = ex;
    atomicAdd(&den[(size_t)c * HEADS + hh], ex);
}

__global__ void hgt_accum_kernel(const float* __restrict__ s, const float* __restrict__ den,
                                 const float* __restrict__ vm,
                                 const int* __restrict__ rows, const int* __restrict__ cols,
                                 float* __restrict__ out, int E)
{
    long long t = (long long)blockIdx.x * blockDim.x + threadIdx.x;
    if (t >= (long long)E * HID) return;
    int e = (int)(t >> 7), f = (int)(t & 127), hh = f >> 5;
    int r = rows[e], c = cols[e];
    float alpha = s[(size_t)e * HEADS + hh] / fmaxf(den[(size_t)c * HEADS + hh], 1e-16f);
    atomicAdd(&out[(size_t)c * HID + f], vm[(size_t)r * HID + f] * alpha);
}

// ---------------------------------------------------------------------------
// Elementwise: exact GELU, sigmoid skip blend, classifier head
// ---------------------------------------------------------------------------
__global__ void gelu_kernel(const float* __restrict__ x, float* __restrict__ y, long long n)
{
    long long t = (long long)blockIdx.x * blockDim.x + threadIdx.x;
    if (t >= n) return;
    float v = x[t];
    y[t] = 0.5f * v * (1.0f + erff(v * 0.70710678118654752f));
}

__global__ void blend_kernel(const float* __restrict__ o, const float* __restrict__ h,
                             const float* __restrict__ skip, float* __restrict__ h2, long long n)
{
    long long t = (long long)blockIdx.x * blockDim.x + threadIdx.x;
    if (t >= n) return;
    float a = 1.0f / (1.0f + expf(-skip[0]));
    h2[t] = a * o[t] + (1.0f - a) * h[t];
}

__global__ void cls_head_kernel(const float* __restrict__ z, const float* __restrict__ W2,
                                const float* __restrict__ b2, float* __restrict__ logits, int M)
{
    int t = blockIdx.x * blockDim.x + threadIdx.x;
    if (t >= M) return;
    const float* zp = z + (size_t)t * 64;
    float acc = b2[0];
#pragma unroll
    for (int j = 0; j < 64; ++j) acc += zp[j] * W2[j];
    logits[t] = acc;
}

// ---------------------------------------------------------------------------
// Host orchestration
// ---------------------------------------------------------------------------
extern "C" void kernel_launch(void* const* d_in, const int* in_sizes, int n_in,
                              void* d_out, int out_size, void* d_ws, size_t ws_size,
                              hipStream_t stream)
{
    (void)in_sizes; (void)n_in; (void)out_size; (void)ws_size;
    const int NC = 200000, NU = 100000, NP = 20000, E = 400000;
    const int Nn[3]  = {NC, NU, NP};
    const int FIN[3] = {32, 16, 16};
    // ET order: (Customer->Claim), (Claim->Customer), (Provider->Claim), (Claim->Provider)
    const int esrc[4] = {1, 0, 2, 0};
    const int edst[4] = {0, 1, 0, 2};

    int gi = 0;
    const float* xIn[3];
    for (int i = 0; i < 3; ++i) xIn[i] = (const float*)d_in[gi++];
    const int* edge[4];
    for (int i = 0; i < 4; ++i) edge[i] = (const int*)d_in[gi++];
    const float *projW[3], *projB[3];
    for (int i = 0; i < 3; ++i) { projW[i] = (const float*)d_in[gi++]; projB[i] = (const float*)d_in[gi++]; }
    const float *sageWl[4], *sageBl[4], *sageWr[4];
    for (int i = 0; i < 4; ++i) {
        sageWl[i] = (const float*)d_in[gi++];
        sageBl[i] = (const float*)d_in[gi++];
        sageWr[i] = (const float*)d_in[gi++];
    }
    const float *bn1g[3], *bn1bp[3];
    for (int i = 0; i < 3; ++i) { bn1g[i] = (const float*)d_in[gi++]; bn1bp[i] = (const float*)d_in[gi++]; }
    const float *kW[3], *kBias[3], *qW[3], *qBias[3], *vW[3], *vBias[3];
    for (int i = 0; i < 3; ++i) { kW[i] = (const float*)d_in[gi++]; kBias[i] = (const float*)d_in[gi++]; }
    for (int i = 0; i < 3; ++i) { qW[i] = (const float*)d_in[gi++]; qBias[i] = (const float*)d_in[gi++]; }
    for (int i = 0; i < 3; ++i) { vW[i] = (const float*)d_in[gi++]; vBias[i] = (const float*)d_in[gi++]; }
    const float *arel[4], *mrel[4], *prel[4];
    for (int i = 0; i < 4; ++i) arel[i] = (const float*)d_in[gi++];
    for (int i = 0; i < 4; ++i) mrel[i] = (const float*)d_in[gi++];
    for (int i = 0; i < 4; ++i) prel[i] = (const float*)d_in[gi++];
    const float *alinW[3], *alinB[3];
    for (int i = 0; i < 3; ++i) { alinW[i] = (const float*)d_in[gi++]; alinB[i] = (const float*)d_in[gi++]; }
    const float* skip[3];
    for (int i = 0; i < 3; ++i) skip[i] = (const float*)d_in[gi++];
    const float *bn2g[3], *bn2bp[3];
    for (int i = 0; i < 3; ++i) { bn2g[i] = (const float*)d_in[gi++]; bn2bp[i] = (const float*)d_in[gi++]; }
    const float* clsW1 = (const float*)d_in[gi++];
    const float* clsB1 = (const float*)d_in[gi++];
    const float* clsW2 = (const float*)d_in[gi++];
    const float* clsB2 = (const float*)d_in[gi++];

    // ---- workspace carve ----
    float* w = (float*)d_ws;
    size_t off = 0;
    auto carve = [&](size_t n) -> float* {
        float* p = w + off;
        off += (n + 63) & ~(size_t)63;
        return p;
    };
    float *hB[3], *aggB[3], *bufK[3], *bufQ[3], *bufV[3];
    for (int i = 0; i < 3; ++i) hB[i]   = carve((size_t)Nn[i] * HID);
    for (int i = 0; i < 3; ++i) aggB[i] = carve((size_t)Nn[i] * HID);
    for (int i = 0; i < 3; ++i) bufK[i] = carve((size_t)Nn[i] * HID);
    for (int i = 0; i < 3; ++i) bufQ[i] = carve((size_t)Nn[i] * HID);
    for (int i = 0; i < 3; ++i) bufV[i] = carve((size_t)Nn[i] * HID);
    float* tmp1 = carve((size_t)NC * HID);
    float* tmp2 = carve((size_t)NC * HID);
    float* sbuf[4];
    for (int i = 0; i < 4; ++i) sbuf[i] = carve((size_t)E * HEADS);
    unsigned* smaxB[3];
    float* denB[3];
    for (int i = 0; i < 3; ++i) smaxB[i] = (unsigned*)carve((size_t)Nn[i] * HEADS);
    for (int i = 0; i < 3; ++i) denB[i]  = carve((size_t)Nn[i] * HEADS);
    float* cnt     = carve(NC);
    float* meanBuf = carve(HID);
    float* varBuf  = carve(HID);
    float* outB[3] = {aggB[0], aggB[1], aggB[2]}; // agg is dead after BN1

    float* out_f = (float*)d_out;
    float* ho[3] = { out_f,
                     out_f + (size_t)NC * HID,
                     out_f + (size_t)NC * HID + (size_t)NU * HID };
    float* logits = out_f + (size_t)(NC + NU + NP) * HID;

    auto gemm = [&](const float* A, int lda, const float* B, int ldb,
                    const float* bias, const float* add, float* C, int ldc,
                    int M, int N, int K, int act) {
        dim3 g((M + 63) / 64, (N + 63) / 64);
        gemm_wmma_kernel<<<g, 256, 0, stream>>>(A, lda, B, ldb, bias, add, C, ldc, M, N, K, act);
    };

    // ---- stage 1: input projection + relu ----
    for (int i = 0; i < 3; ++i)
        gemm(xIn[i], FIN[i], projW[i], HID, projB[i], nullptr, hB[i], HID,
             Nn[i], HID, FIN[i], 1);

    // ---- stage 2: SAGE hetero conv ----
    for (int i = 0; i < 3; ++i)
        (void)hipMemsetAsync(aggB[i], 0, (size_t)Nn[i] * HID * sizeof(float), stream);
    for (int et = 0; et < 4; ++et) {
        int si = esrc[et], di = edst[et], Nd = Nn[di];
        (void)hipMemsetAsync(tmp1, 0, (size_t)Nd * HID * sizeof(float), stream);
        (void)hipMemsetAsync(cnt, 0, (size_t)Nd * sizeof(float), stream);
        long long ntEF = (long long)E * HID;
        scatter_add_kernel<<<(int)((ntEF + 255) / 256), 256, 0, stream>>>(
            hB[si], edge[et], edge[et] + E, tmp1, E);
        count_kernel<<<(E + 255) / 256, 256, 0, stream>>>(edge[et] + E, cnt, E);
        div_cnt_kernel<<<(Nd * HID + 255) / 256, 256, 0, stream>>>(tmp1, cnt, Nd);
        gemm(tmp1, HID, sageWl[et], HID, sageBl[et], nullptr, tmp2, HID, Nd, HID, HID, 0);
        gemm(hB[di], HID, sageWr[et], HID, nullptr, tmp2, tmp2, HID, Nd, HID, HID, 0);
        rownorm_acc_kernel<<<(Nd + 3) / 4, 128, 0, stream>>>(tmp2, aggB[di], Nd);
    }

    // ---- stage 3: bn1 + relu (agg -> h, overwrite) ----
    for (int i = 0; i < 3; ++i) {
        bn_stats_kernel<<<HID, 256, 0, stream>>>(aggB[i], Nn[i], meanBuf, varBuf);
        long long n = (long long)Nn[i] * HID;
        bn_apply_relu_kernel<<<(int)((n + 255) / 256), 256, 0, stream>>>(
            aggB[i], meanBuf, varBuf, bn1g[i], bn1bp[i], hB[i], Nn[i]);
    }

    // ---- stage 4: k/q/v projections ----
    for (int i = 0; i < 3; ++i) {
        gemm(hB[i], HID, kW[i], HID, kBias[i], nullptr, bufK[i], HID, Nn[i], HID, HID, 0);
        gemm(hB[i], HID, qW[i], HID, qBias[i], nullptr, bufQ[i], HID, Nn[i], HID, HID, 0);
        gemm(hB[i], HID, vW[i], HID, vBias[i], nullptr, bufV[i], HID, Nn[i], HID, HID, 0);
    }

    // ---- stage 5: HGT attention (3-pass segment softmax) ----
    for (int i = 0; i < 3; ++i) {
        int n = Nn[i] * HEADS;
        init_smax_kernel<<<(n + 255) / 256, 256, 0, stream>>>(smaxB[i], n);
        (void)hipMemsetAsync(denB[i], 0, (size_t)n * sizeof(float), stream);
        (void)hipMemsetAsync(outB[i], 0, (size_t)Nn[i] * HID * sizeof(float), stream);
    }
    // pass A: ka = k_src @ a_rel (per-head 32x32 GEMM), scores + segment max
    for (int et = 0; et < 4; ++et) {
        int si = esrc[et], di = edst[et];
        for (int hh = 0; hh < HEADS; ++hh)
            gemm(bufK[si] + hh * DH, HID, arel[et] + hh * DH * DH, DH,
                 nullptr, nullptr, tmp1 + hh * DH, HID, Nn[si], DH, DH, 0);
        int nthr = E * HEADS;
        hgt_score_kernel<<<(nthr + 255) / 256, 256, 0, stream>>>(
            bufQ[di], tmp1, edge[et], edge[et] + E, prel[et], sbuf[et], smaxB[di], E);
    }
    // pass B: exp + segment sum
    for (int et = 0; et < 4; ++et) {
        int di = edst[et], nthr = E * HEADS;
        hgt_expsum_kernel<<<(nthr + 255) / 256, 256, 0, stream>>>(
            sbuf[et], smaxB[di], edge[et] + E, denB[di], E);
    }
    // pass C: vm = v_src @ m_rel, weighted accumulate
    for (int et = 0; et < 4; ++et) {
        int si = esrc[et], di = edst[et];
        for (int hh = 0; hh < HEADS; ++hh)
            gemm(bufV[si] + hh * DH, HID, mrel[et] + hh * DH * DH, DH,
                 nullptr, nullptr, tmp1 + hh * DH, HID, Nn[si], DH, DH, 0);
        long long ntEF = (long long)E * HID;
        hgt_accum_kernel<<<(int)((ntEF + 255) / 256), 256, 0, stream>>>(
            sbuf[et], denB[di], tmp1, edge[et], edge[et] + E, outB[di], E);
    }

    // ---- stage 6: gelu -> a_lin -> skip blend ----
    for (int i = 0; i < 3; ++i) {
        long long n = (long long)Nn[i] * HID;
        gelu_kernel<<<(int)((n + 255) / 256), 256, 0, stream>>>(outB[i], tmp1, n);
        gemm(tmp1, HID, alinW[i], HID, alinB[i], nullptr, tmp2, HID, Nn[i], HID, HID, 0);
        blend_kernel<<<(int)((n + 255) / 256), 256, 0, stream>>>(tmp2, hB[i], skip[i], outB[i], n);
    }

    // ---- stage 7: bn2 + relu -> final h (straight into d_out) ----
    for (int i = 0; i < 3; ++i) {
        bn_stats_kernel<<<HID, 256, 0, stream>>>(outB[i], Nn[i], meanBuf, varBuf);
        long long n = (long long)Nn[i] * HID;
        bn_apply_relu_kernel<<<(int)((n + 255) / 256), 256, 0, stream>>>(
            outB[i], meanBuf, varBuf, bn2g[i], bn2bp[i], ho[i], Nn[i]);
    }

    // ---- stage 8: classifier on Claim ----
    gemm(ho[0], HID, clsW1, 64, clsB1, nullptr, tmp1, 64, NC, 64, HID, 1);
    cls_head_kernel<<<(NC + 255) / 256, 256, 0, stream>>>(tmp1, clsW2, clsB2, logits, NC);
}
